// MoEPackageLayer_2164663517797
// MI455X (gfx1250) — matrix operand order repaired and checked by reading, not compiled, not run on hardware
//
#include <hip/hip_runtime.h>
#include <math.h>
#include <stddef.h>

// MoE: T=4096 tokens, H=2048 d_model, F=4096 d_ffn, E=8 experts, top-2.
// Gather tokens per expert -> grouped GEMMs in bf16 WMMA with f32 accumulate.
#define T_TOK 4096
#define HDIM  2048
#define FDIM  4096
#define NEXP  8

typedef __attribute__((ext_vector_type(16))) __bf16 v16bf;
typedef __attribute__((ext_vector_type(8)))  __bf16 v8bf;
typedef __attribute__((ext_vector_type(8)))  float  v8f;

// -------- workspace layout (bytes); needs ~82 MiB --------
#define OFF_COUNTS 0u
#define OFF_BASE   256u
#define OFF_TIDS   512u                               // E*T ints   = 128 KiB
#define OFF_TW     (512u + (unsigned)(NEXP*T_TOK)*4u) // E*T floats = 128 KiB
#define OFF_XBF    (1u << 20)                         // T*H bf16   = 16 MiB
#define OFF_ACT    (18u << 20)                        // 8192*F bf16= 64 MiB

#define BM 128
#define BN 64
#define BK 32
#define LDST 40   // LDS row stride in halfs (80 B = 20 banks; 16B-aligned rows)

// Build a 16-half fragment from two contiguous 16-byte LDS runs.
static __device__ __forceinline__ v16bf frag_cat(const __bf16* lo, const __bf16* hi) {
  const v8bf a = *(const v8bf*)lo;
  const v8bf b = *(const v8bf*)hi;
  return __builtin_shufflevector(a, b, 0, 1, 2, 3, 4, 5, 6, 7,
                                 8, 9, 10, 11, 12, 13, 14, 15);
}

// A-matrix 16x32 bf16 (ISA 7.12.2): lane=hs*16+M; halfs 0..7 = row M cols
// [hs*8, hs*8+8), halfs 8..15 = cols [16+hs*8, +8).
static __device__ __forceinline__ v16bf load_fragA(const __bf16* p, int lane) {
  const int M = lane & 15, hs = lane >> 4;
  const __bf16* r = p + M * LDST;
  return frag_cat(r + hs * 8, r + 16 + hs * 8);
}
// B-matrix 32x16 bf16 stored N-MAJOR (transposed) in LDS: row n holds K 0..31.
// lane=hs*16+N; halfs 0..7 = BsT[N][hs*8 .. hs*8+7], halfs 8..15 = BsT[N][16+hs*8 .. +8).
static __device__ __forceinline__ v16bf load_fragB(const __bf16* p, int lane) {
  const int N = lane & 15, hs = lane >> 4;
  const __bf16* r = p + N * LDST;
  return frag_cat(r + hs * 8, r + 16 + hs * 8);
}

#define WMMA_BF16(A, B, C) \
  __builtin_amdgcn_wmma_f32_16x16x32_bf16(false, (A), false, (B), (short)0, (C), false, false)

// Async global->LDS copy of 32 contiguous bytes (two B128 transfers, ASYNCcnt).
// offset: applies to both LDS and global sides (ISA 08_async_tensor 4.4).
static __device__ __forceinline__ void async_copy32(__bf16* dst_lds,
                                                    const __bf16* src_glb) {
  const unsigned lds = (unsigned)(size_t)dst_lds;  // addrspace(3) byte offset
  asm volatile("global_load_async_to_lds_b128 %0, %1, off\n\t"
               "global_load_async_to_lds_b128 %0, %1, off offset:16"
               :
               : "v"(lds), "v"(src_glb)
               : "memory");
}
static __device__ __forceinline__ void wait_async0() {
  asm volatile("s_wait_asynccnt 0x0" ::: "memory");
}

static __device__ __forceinline__ void zero16h(__bf16* dst) {
  const v8bf z = {};
  *(v8bf*)dst = z;
  *(v8bf*)(dst + 8) = z;
}
// Transpose-stage 8 fp32 (fixed n, k..k+7, global row stride `gs`) -> one 16B store.
static __device__ __forceinline__ void stage_bT8(__bf16* dst, const float* src, int gs) {
  v8bf v;
#pragma unroll
  for (int i = 0; i < 8; ++i) v[i] = (__bf16)src[(size_t)i * gs];
  *(v8bf*)dst = v;
}

// ---------------- small utility kernels ----------------
__global__ void k_init(int* counts) {
  if (threadIdx.x < NEXP) counts[threadIdx.x] = 0;
}

__global__ void k_zero4(float4* p, int n4) {
  int i = blockIdx.x * blockDim.x + threadIdx.x;
  if (i < n4) p[i] = make_float4(0.f, 0.f, 0.f, 0.f);
}

__global__ void k_cvtx(const float* __restrict__ x, __bf16* __restrict__ xbf, int n8) {
  int i = blockIdx.x * blockDim.x + threadIdx.x;
  if (i < n8) {
    const float4 a = reinterpret_cast<const float4*>(x)[2 * i + 0];
    const float4 b = reinterpret_cast<const float4*>(x)[2 * i + 1];
    v8bf v;
    v[0] = (__bf16)a.x; v[1] = (__bf16)a.y; v[2] = (__bf16)a.z; v[3] = (__bf16)a.w;
    v[4] = (__bf16)b.x; v[5] = (__bf16)b.y; v[6] = (__bf16)b.z; v[7] = (__bf16)b.w;
    reinterpret_cast<v8bf*>(xbf)[i] = v;
  }
}

__global__ void k_prefix(const int* __restrict__ counts, int* __restrict__ base) {
  if (threadIdx.x == 0 && blockIdx.x == 0) {
    int run = 0;
#pragma unroll
    for (int e = 0; e < NEXP; ++e) { base[e] = run; run += counts[e]; }
  }
}

// ---------------- gating: one wave32 per token ----------------
__global__ __launch_bounds__(256) void k_gate(const float* __restrict__ x,
                                              const float* __restrict__ gw,
                                              int* __restrict__ counts,
                                              int* __restrict__ tids,
                                              float* __restrict__ tw) {
  const int wv = threadIdx.x >> 5, lane = threadIdx.x & 31;
  const int t = blockIdx.x * 8 + wv;
  if (t >= T_TOK) return;

  float acc[NEXP];
#pragma unroll
  for (int e = 0; e < NEXP; ++e) acc[e] = 0.f;

  const float4* xr = reinterpret_cast<const float4*>(x + (size_t)t * HDIM);
  const float4* gw4 = reinterpret_cast<const float4*>(gw);
  for (int q = lane; q < HDIM / 4; q += 32) {
    const float4 xv = xr[q];
#pragma unroll
    for (int e = 0; e < NEXP; ++e) {
      const float4 g = gw4[e * (HDIM / 4) + q];
      acc[e] += xv.x * g.x + xv.y * g.y + xv.z * g.z + xv.w * g.w;
    }
  }
#pragma unroll
  for (int off = 16; off > 0; off >>= 1)
#pragma unroll
    for (int e = 0; e < NEXP; ++e) acc[e] += __shfl_xor(acc[e], off, 32);

  if (lane == 0) {
    int i0 = 0;
#pragma unroll
    for (int e = 1; e < NEXP; ++e) if (acc[e] > acc[i0]) i0 = e;
    int i1 = -1;
#pragma unroll
    for (int e = 0; e < NEXP; ++e)
      if (e != i0 && (i1 < 0 || acc[e] > acc[i1])) i1 = e;
    const float l0 = acc[i0], l1 = acc[i1];
    const float e1 = __expf(l1 - l0);        // l0 >= l1
    const float inv = 1.f / (1.f + e1);
    int p0 = atomicAdd(&counts[i0], 1);
    tids[i0 * T_TOK + p0] = t; tw[i0 * T_TOK + p0] = inv;
    int p1 = atomicAdd(&counts[i1], 1);
    tids[i1 * T_TOK + p1] = t; tw[i1 * T_TOK + p1] = e1 * inv;
  }
}

// ---------------- stage 1: act = silu(Xg @ W1a) * (Xg @ W1b) ----------------
// block = 256 threads (8 waves, 4(M) x 2(N)); tile 128 tokens x 64 ffn cols.
__global__ __launch_bounds__(256) void k_stage1(const __bf16* __restrict__ xbf,
                                                const float* __restrict__ w1,
                                                const int* __restrict__ counts,
                                                const int* __restrict__ basep,
                                                const int* __restrict__ tids,
                                                __bf16* __restrict__ act) {
  const int e = blockIdx.x, tm = blockIdx.y, tn = blockIdx.z;
  const int cnt = counts[e];
  if (tm * BM >= cnt) return;
  const int base = basep[e];
  const int tid = threadIdx.x, lane = tid & 31, wv = tid >> 5;
  const int wm = wv & 3, wn = wv >> 2;
  const int n0 = tn * BN;

  __shared__ __bf16 As[BM][LDST];     // row-major, K contiguous
  __shared__ __bf16 BaT[BN][LDST];    // N-major (transposed), K contiguous
  __shared__ __bf16 BbT[BN][LDST];
  __shared__ int rowtok[BM];

  if (tid < BM) {
    const int m = tm * BM + tid;
    rowtok[tid] = (m < cnt) ? tids[e * T_TOK + m] : -1;
  }
  __syncthreads();

  v8f accA[2][2], accB[2][2];
  const v8f vz = {};
#pragma unroll
  for (int i = 0; i < 2; ++i)
#pragma unroll
    for (int j = 0; j < 2; ++j) { accA[i][j] = vz; accB[i][j] = vz; }

  const float* w1e = w1 + (size_t)e * HDIM * (2 * FDIM);
  const int ar = tid >> 1, aseg = tid & 1;  // A staging: row + 16-half chunk
  const int atok = rowtok[ar];
  const int bn = tid & 63, bkg = tid >> 6;  // B staging: column n, k-group

  for (int ks = 0; ks < HDIM; ks += BK) {
    if (atok >= 0)
      async_copy32(&As[ar][aseg * 16],
                   xbf + (size_t)atok * HDIM + ks + aseg * 16);  // ASYNCcnt path
    else
      zero16h(&As[ar][aseg * 16]);

    {
      const float* srcA = w1e + (size_t)(ks + bkg * 8) * (2 * FDIM) + n0 + bn;
      stage_bT8(&BaT[bn][bkg * 8], srcA, 2 * FDIM);
      stage_bT8(&BbT[bn][bkg * 8], srcA + FDIM, 2 * FDIM);
      if (ks + BK < HDIM)
        __builtin_prefetch(srcA + (size_t)BK * (2 * FDIM), 0, 0); // global_prefetch_b8
    }
    wait_async0();
    __syncthreads();

    const v16bf a0 = load_fragA(&As[wm * 32 + 0][0], lane);
    const v16bf a1 = load_fragA(&As[wm * 32 + 16][0], lane);
    const v16bf ba0 = load_fragB(&BaT[wn * 32 + 0][0], lane);
    const v16bf ba1 = load_fragB(&BaT[wn * 32 + 16][0], lane);
    const v16bf bb0 = load_fragB(&BbT[wn * 32 + 0][0], lane);
    const v16bf bb1 = load_fragB(&BbT[wn * 32 + 16][0], lane);

    accA[0][0] = WMMA_BF16(a0, ba0, accA[0][0]);
    accA[0][1] = WMMA_BF16(a0, ba1, accA[0][1]);
    accA[1][0] = WMMA_BF16(a1, ba0, accA[1][0]);
    accA[1][1] = WMMA_BF16(a1, ba1, accA[1][1]);
    accB[0][0] = WMMA_BF16(a0, bb0, accB[0][0]);
    accB[0][1] = WMMA_BF16(a0, bb1, accB[0][1]);
    accB[1][0] = WMMA_BF16(a1, bb0, accB[1][0]);
    accB[1][1] = WMMA_BF16(a1, bb1, accB[1][1]);
    __syncthreads();
  }

  // epilogue: silu(a)*b -> bf16. C layout: VGPR v -> M=(lane>>4)*8+v, N=lane&15.
#pragma unroll
  for (int i = 0; i < 2; ++i)
#pragma unroll
    for (int j = 0; j < 2; ++j)
#pragma unroll
      for (int v = 0; v < 8; ++v) {
        const int rl = wm * 32 + i * 16 + (lane >> 4) * 8 + v;
        const int m = tm * BM + rl;
        if (m < cnt) {
          const int col = n0 + wn * 32 + j * 16 + (lane & 15);
          const float a = accA[i][j][v], b = accB[i][j][v];
          const float g = a * (1.f / (1.f + __expf(-a))) * b; // silu(a)*b
          act[(size_t)(base + m) * FDIM + col] = (__bf16)g;
        }
      }
}

// ---------------- stage 2: out[token] += w * (act @ W2) ----------------
__global__ __launch_bounds__(256) void k_stage2(const __bf16* __restrict__ act,
                                                const float* __restrict__ w2,
                                                const int* __restrict__ counts,
                                                const int* __restrict__ basep,
                                                const int* __restrict__ tids,
                                                const float* __restrict__ tw,
                                                float* __restrict__ out) {
  const int e = blockIdx.x, tm = blockIdx.y, tn = blockIdx.z;
  const int cnt = counts[e];
  if (tm * BM >= cnt) return;
  const int base = basep[e];
  const int tid = threadIdx.x, lane = tid & 31, wv = tid >> 5;
  const int wm = wv & 3, wn = wv >> 2;
  const int n0 = tn * BN;

  __shared__ __bf16 As[BM][LDST];
  __shared__ __bf16 BsT[BN][LDST];

  v8f acc[2][2];
  const v8f vz = {};
#pragma unroll
  for (int i = 0; i < 2; ++i)
#pragma unroll
    for (int j = 0; j < 2; ++j) acc[i][j] = vz;

  const float* w2e = w2 + (size_t)e * FDIM * HDIM;
  const int ar = tid >> 1, aseg = tid & 1;
  const int am = tm * BM + ar;
  const int bn = tid & 63, bkg = tid >> 6;

  for (int ks = 0; ks < FDIM; ks += BK) {
    if (am < cnt)
      async_copy32(&As[ar][aseg * 16],
                   act + (size_t)(base + am) * FDIM + ks + aseg * 16);  // ASYNCcnt
    else
      zero16h(&As[ar][aseg * 16]);

    {
      const float* src = w2e + (size_t)(ks + bkg * 8) * HDIM + n0 + bn;
      stage_bT8(&BsT[bn][bkg * 8], src, HDIM);
      if (ks + BK < FDIM)
        __builtin_prefetch(src + (size_t)BK * HDIM, 0, 0); // global_prefetch_b8
    }
    wait_async0();
    __syncthreads();

    const v16bf a0 = load_fragA(&As[wm * 32 + 0][0], lane);
    const v16bf a1 = load_fragA(&As[wm * 32 + 16][0], lane);
    const v16bf b0 = load_fragB(&BsT[wn * 32 + 0][0], lane);
    const v16bf b1 = load_fragB(&BsT[wn * 32 + 16][0], lane);

    acc[0][0] = WMMA_BF16(a0, b0, acc[0][0]);
    acc[0][1] = WMMA_BF16(a0, b1, acc[0][1]);
    acc[1][0] = WMMA_BF16(a1, b0, acc[1][0]);
    acc[1][1] = WMMA_BF16(a1, b1, acc[1][1]);
    __syncthreads();
  }

#pragma unroll
  for (int i = 0; i < 2; ++i)
#pragma unroll
    for (int j = 0; j < 2; ++j)
#pragma unroll
      for (int v = 0; v < 8; ++v) {
        const int rl = wm * 32 + i * 16 + (lane >> 4) * 8 + v;
        const int m = tm * BM + rl;
        if (m < cnt) {
          const int token = tids[e * T_TOK + m];
          const float wgt = tw[e * T_TOK + m];
          const int h = n0 + wn * 32 + j * 16 + (lane & 15);
          atomicAdd(&out[(size_t)token * HDIM + h], wgt * acc[i][j][v]);
        }
      }
}

// ---------------- host entry ----------------
extern "C" void kernel_launch(void* const* d_in, const int* in_sizes, int n_in,
                              void* d_out, int out_size, void* d_ws, size_t ws_size,
                              hipStream_t stream) {
  (void)in_sizes; (void)n_in; (void)out_size; (void)ws_size;
  const float* x  = (const float*)d_in[0];  // [T,H]
  const float* gw = (const float*)d_in[1];  // [E,H]
  const float* w1 = (const float*)d_in[2];  // [E,H,2F]
  const float* w2 = (const float*)d_in[3];  // [E,F,H]
  float* out = (float*)d_out;               // [T,H]

  char* ws = (char*)d_ws;
  int*    counts = (int*)(ws + OFF_COUNTS);
  int*    basep  = (int*)(ws + OFF_BASE);
  int*    tids   = (int*)(ws + OFF_TIDS);
  float*  tw     = (float*)(ws + OFF_TW);
  __bf16* xbf    = (__bf16*)(ws + OFF_XBF);
  __bf16* act    = (__bf16*)(ws + OFF_ACT);

  const int n4 = (T_TOK * HDIM) / 4;
  const int n8 = (T_TOK * HDIM) / 8;

  k_init<<<1, 256, 0, stream>>>(counts);
  k_zero4<<<(n4 + 255) / 256, 256, 0, stream>>>((float4*)out, n4);
  k_cvtx<<<(n8 + 255) / 256, 256, 0, stream>>>(x, xbf, n8);
  k_gate<<<T_TOK / 8, 256, 0, stream>>>(x, gw, counts, tids, tw);
  k_prefix<<<1, 32, 0, stream>>>(counts, basep);

  dim3 g1(NEXP, T_TOK / BM, FDIM / BN);  // (8, 32, 64)
  k_stage1<<<g1, 256, 0, stream>>>(xbf, w1, counts, basep, tids, act);

  dim3 g2(NEXP, T_TOK / BM, HDIM / BN);  // (8, 32, 32)
  k_stage2<<<g2, 256, 0, stream>>>(act, w2, counts, basep, tids, tw, out);
}
